// Block_10720238371232
// MI455X (gfx1250) — matrix operand order repaired and checked by reading, not compiled, not run on hardware
//
#include <hip/hip_runtime.h>

// ---- problem constants (match reference) ----
#define NBATCH 4
#define TSEQ   2048
#define CDIM   1024
#define NHEAD  16
#define HDIM   64
#define TOK    (NBATCH * TSEQ)   // 8192

typedef _Float16 h8v  __attribute__((ext_vector_type(8)));
typedef _Float16 h16v __attribute__((ext_vector_type(16)));
typedef float    f8v  __attribute__((ext_vector_type(8)));

__device__ __forceinline__ h8v ld8(const _Float16* p) {
    return *reinterpret_cast<const h8v*>(p);
}
__device__ __forceinline__ h16v mk16(h8v lo, h8v hi) {
    h16v r;
#pragma unroll
    for (int i = 0; i < 8; ++i) { r[i] = lo[i]; r[i + 8] = hi[i]; }
    return r;
}
__device__ __forceinline__ f8v wmma16(h16v a, h16v b, f8v c) {
    return __builtin_amdgcn_wmma_f32_16x16x32_f16(false, a, false, b, (short)0, c, false, false);
}

// gfx1250 async copy: global -> LDS without a VGPR round trip (ASYNCcnt path).
__device__ __forceinline__ void async_g2l_b128(void* lds, const void* gptr) {
    unsigned int       laddr = (unsigned int)(uintptr_t)lds;       // LDS offset = addr[31:0]
    unsigned long long gaddr = (unsigned long long)(uintptr_t)gptr;
    asm volatile("global_load_async_to_lds_b128 %0, %1, off"
                 :: "v"(laddr), "v"(gaddr) : "memory");
}
__device__ __forceinline__ void wait_asynccnt0() {
    asm volatile("s_wait_asynccnt 0x0" ::: "memory");
}

// ---------------------------------------------------------------------------
// fp32 -> f16 weight conversion
// ---------------------------------------------------------------------------
__global__ void f32_to_f16_kernel(const float* __restrict__ in,
                                  _Float16* __restrict__ out, int n) {
    int i = blockIdx.x * 256 + threadIdx.x;
    if (i < n) out[i] = (_Float16)in[i];
}

// ---------------------------------------------------------------------------
// LayerNorm (one block per token row of 1024), f32 in -> f16 out
// ---------------------------------------------------------------------------
__global__ __launch_bounds__(256)
void ln_f16_kernel(const float* __restrict__ x, const float* __restrict__ g,
                   const float* __restrict__ bta, _Float16* __restrict__ out) {
    __shared__ float red[256];
    __shared__ float smean, srstd;
    const int row = blockIdx.x, tid = threadIdx.x;
    const float* xr = x + (size_t)row * CDIM;
    float v[4];
    float s = 0.f;
#pragma unroll
    for (int j = 0; j < 4; ++j) { v[j] = xr[tid + 256 * j]; s += v[j]; }
    red[tid] = s; __syncthreads();
    for (int off = 128; off > 0; off >>= 1) {
        if (tid < off) red[tid] += red[tid + off];
        __syncthreads();
    }
    if (tid == 0) smean = red[0] * (1.0f / CDIM);
    __syncthreads();
    float mu = smean, s2 = 0.f;
#pragma unroll
    for (int j = 0; j < 4; ++j) { float d = v[j] - mu; s2 += d * d; }
    red[tid] = s2; __syncthreads();
    for (int off = 128; off > 0; off >>= 1) {
        if (tid < off) red[tid] += red[tid + off];
        __syncthreads();
    }
    if (tid == 0) srstd = rsqrtf(red[0] * (1.0f / CDIM) + 1e-5f);
    __syncthreads();
    float rs = srstd;
#pragma unroll
    for (int j = 0; j < 4; ++j) {
        int c = tid + 256 * j;
        out[(size_t)row * CDIM + c] = (_Float16)((v[j] - mu) * rs * g[c] + bta[c]);
    }
}

// ---------------------------------------------------------------------------
// WMMA GEMM: D[m,n] = sum_k A[m,k] * W[n,k] + bias[n]
// A: MxK f16 row-major, W: NxK f16 row-major (computes A @ W^T).
// 256 threads = 8 waves; block tile 256(M) x 64(N); each wave 32(M) x 64(N):
// two A fragments share each B fragment -> 8 WMMAs per K-step per wave.
// B tile staged via global_load_async_to_lds_b128 (ASYNCcnt).
// mode 0: out32 = res + D        (residual add, fp32)
// mode 1: QKV scatter: q*0.125 -> [b,h,t,d]; k -> [b,h,t,d]; v -> V^T [b,h,d,t]
// mode 2: out16 = gelu(D)        (f16)
// ---------------------------------------------------------------------------
__global__ __launch_bounds__(256)
void gemm_wmma_kernel(const _Float16* __restrict__ A, const _Float16* __restrict__ W,
                      const float* __restrict__ bias, int M, int N, int K, int mode,
                      const float* __restrict__ res, float* __restrict__ out32,
                      _Float16* __restrict__ out16) {
    __shared__ _Float16 Bs[64 * 40];   // 64 rows (n) x 32 k, padded stride 40
    const int tid  = threadIdx.x;
    const int wave = tid >> 5, lane = tid & 31;
    const int halfid = lane >> 4, l16 = lane & 15;
    const int n0 = blockIdx.x * 64;
    const int m0 = blockIdx.y * 256;

    const int arow = m0 + wave * 32 + l16;
    const _Float16* Ar0 = A + (size_t)arow * K;
    const _Float16* Ar1 = Ar0 + (size_t)16 * K;

    f8v zero = {};
    f8v acc[2][4];
#pragma unroll
    for (int af = 0; af < 2; ++af)
#pragma unroll
        for (int t = 0; t < 4; ++t) acc[af][t] = zero;

    const int brow = tid >> 2, bch = tid & 3;     // 64 rows x 4 chunks of 8 halves
    const _Float16* Wp = W + (size_t)(n0 + brow) * K + bch * 8;
    _Float16* lds_dst = &Bs[brow * 40 + bch * 8];

    for (int kb = 0; kb < K; kb += 32) {
        async_g2l_b128(lds_dst, Wp + kb);             // B tile -> LDS (async)
        if (kb + 32 < K) __builtin_prefetch(Wp + kb + 32, 0, 1);
        wait_asynccnt0();
        __syncthreads();
        // A fragments: rows l16 and l16+16 of the wave strip
        h16v av0 = mk16(ld8(Ar0 + kb + halfid * 16), ld8(Ar0 + kb + halfid * 16 + 8));
        h16v av1 = mk16(ld8(Ar1 + kb + halfid * 16), ld8(Ar1 + kb + halfid * 16 + 8));
#pragma unroll
        for (int nt = 0; nt < 4; ++nt) {
            const _Float16* bp = &Bs[(nt * 16 + l16) * 40 + halfid * 16];
            h16v bv = mk16(ld8(bp), ld8(bp + 8));
            acc[0][nt] = wmma16(av0, bv, acc[0][nt]);
            acc[1][nt] = wmma16(av1, bv, acc[1][nt]);
        }
        __syncthreads();
    }

    float bn[4];
#pragma unroll
    for (int nt = 0; nt < 4; ++nt) bn[nt] = bias[n0 + nt * 16 + l16];

    if (mode == 0) {                       // residual add -> fp32
#pragma unroll
        for (int af = 0; af < 2; ++af)
#pragma unroll
            for (int nt = 0; nt < 4; ++nt) {
                int n = n0 + nt * 16 + l16;
#pragma unroll
                for (int i = 0; i < 8; ++i) {
                    int rowm = m0 + wave * 32 + af * 16 + i + 8 * halfid;
                    size_t idx = (size_t)rowm * N + n;
                    out32[idx] = res[idx] + acc[af][nt][i] + bn[nt];
                }
            }
    } else if (mode == 2) {                // exact GELU -> f16
#pragma unroll
        for (int af = 0; af < 2; ++af)
#pragma unroll
            for (int nt = 0; nt < 4; ++nt) {
                int n = n0 + nt * 16 + l16;
#pragma unroll
                for (int i = 0; i < 8; ++i) {
                    int rowm = m0 + wave * 32 + af * 16 + i + 8 * halfid;
                    float v = acc[af][nt][i] + bn[nt];
                    float gl = 0.5f * v * (1.0f + erff(v * 0.70710678118f));
                    out16[(size_t)rowm * N + n] = (_Float16)gl;
                }
            }
    } else {                               // mode 1: QKV scatter
        _Float16* qb  = out16;
        _Float16* kb2 = out16 + (size_t)TOK * CDIM;
        _Float16* vtb = out16 + (size_t)2 * TOK * CDIM;
#pragma unroll
        for (int af = 0; af < 2; ++af)
#pragma unroll
            for (int nt = 0; nt < 4; ++nt) {
                int n = n0 + nt * 16 + l16;
                int seg = n >> 10;         // 0=q 1=k 2=v (n0 blocks never straddle)
                int c = n & 1023;
                int hh = c >> 6, dd = c & 63;
#pragma unroll
                for (int i = 0; i < 8; ++i) {
                    int rowm = m0 + wave * 32 + af * 16 + i + 8 * halfid;
                    int bb = rowm >> 11, tt = rowm & 2047;
                    float v = acc[af][nt][i] + bn[nt];
                    if (seg == 0)
                        qb[(((size_t)(bb * NHEAD + hh) * TSEQ) + tt) * HDIM + dd] =
                            (_Float16)(v * 0.125f);          // fold 1/sqrt(64)
                    else if (seg == 1)
                        kb2[(((size_t)(bb * NHEAD + hh) * TSEQ) + tt) * HDIM + dd] = (_Float16)v;
                    else
                        vtb[(((size_t)(bb * NHEAD + hh) * HDIM) + dd) * TSEQ + tt] = (_Float16)v;
                }
            }
    }
}

// ---------------------------------------------------------------------------
// Flash-style attention. grid = B*H*(T/128); 8 waves, each owns 16 query rows.
// q pre-scaled; scores via WMMA; online softmax; P restriped via per-wave LDS
// patch; y += P @ V using V^T layout. Output f16 [B,T,C] for the proj GEMM.
// ---------------------------------------------------------------------------
__global__ __launch_bounds__(256)
void attn_wmma_kernel(const _Float16* __restrict__ qf, const _Float16* __restrict__ kf,
                      const _Float16* __restrict__ vt, _Float16* __restrict__ yout) {
    __shared__ _Float16 pbuf[8][16][40];   // per-wave 16x32 P tile, padded
    const int tid  = threadIdx.x;
    const int wave = tid >> 5, lane = tid & 31;
    const int halfid = lane >> 4, l16 = lane & 15;

    const int blk  = blockIdx.x;
    const int qblk = blk & 15;
    const int h    = (blk >> 4) & 15;
    const int b    = blk >> 8;
    const size_t bh = (size_t)(b * NHEAD + h);
    const int row0 = qblk * 128 + wave * 16;

    const _Float16* qrow = qf + (bh * TSEQ + row0 + l16) * HDIM;
    h16v aq0 = mk16(ld8(qrow + halfid * 16),      ld8(qrow + halfid * 16 + 8));
    h16v aq1 = mk16(ld8(qrow + 32 + halfid * 16), ld8(qrow + 32 + halfid * 16 + 8));

    f8v zero = {};
    f8v vy[4];
#pragma unroll
    for (int t = 0; t < 4; ++t) vy[t] = zero;
    float m_i[8], l_i[8];
#pragma unroll
    for (int i = 0; i < 8; ++i) { m_i[i] = -3.0e38f; l_i[i] = 0.f; }

    for (int kt = 0; kt < TSEQ; kt += 32) {
        const _Float16* k0 = kf + (bh * TSEQ + kt + l16) * HDIM;
        const _Float16* k1 = kf + (bh * TSEQ + kt + 16 + l16) * HDIM;
        f8v s0 = zero, s1 = zero;
        {
            h16v b00 = mk16(ld8(k0 + halfid * 16),      ld8(k0 + halfid * 16 + 8));
            h16v b01 = mk16(ld8(k0 + 32 + halfid * 16), ld8(k0 + 32 + halfid * 16 + 8));
            s0 = wmma16(aq0, b00, s0);
            s0 = wmma16(aq1, b01, s0);
            h16v b10 = mk16(ld8(k1 + halfid * 16),      ld8(k1 + halfid * 16 + 8));
            h16v b11 = mk16(ld8(k1 + 32 + halfid * 16), ld8(k1 + 32 + halfid * 16 + 8));
            s1 = wmma16(aq0, b10, s1);
            s1 = wmma16(aq1, b11, s1);
        }
        // ---- online softmax (rows split across the two half-waves) ----
        float tm[8];
#pragma unroll
        for (int i = 0; i < 8; ++i) tm[i] = fmaxf(s0[i], s1[i]);
#pragma unroll
        for (int msk = 1; msk < 16; msk <<= 1)
#pragma unroll
            for (int i = 0; i < 8; ++i) tm[i] = fmaxf(tm[i], __shfl_xor(tm[i], msk, 16));

        float p0[8], p1[8], rsum[8], alpha[8];
#pragma unroll
        for (int i = 0; i < 8; ++i) {
            float mnew = fmaxf(m_i[i], tm[i]);
            alpha[i] = __expf(m_i[i] - mnew);
            m_i[i] = mnew;
            p0[i] = __expf(s0[i] - mnew);
            p1[i] = __expf(s1[i] - mnew);
            rsum[i] = p0[i] + p1[i];
        }
#pragma unroll
        for (int msk = 1; msk < 16; msk <<= 1)
#pragma unroll
            for (int i = 0; i < 8; ++i) rsum[i] += __shfl_xor(rsum[i], msk, 16);
#pragma unroll
        for (int i = 0; i < 8; ++i) l_i[i] = l_i[i] * alpha[i] + rsum[i];
#pragma unroll
        for (int t = 0; t < 4; ++t)
#pragma unroll
            for (int i = 0; i < 8; ++i) vy[t][i] *= alpha[i];

        // ---- restripe P (D-layout) into an A fragment via per-wave LDS ----
#pragma unroll
        for (int i = 0; i < 8; ++i) {
            int r = i + 8 * halfid;
            pbuf[wave][r][l16]      = (_Float16)p0[i];
            pbuf[wave][r][16 + l16] = (_Float16)p1[i];
        }
        const _Float16* prow = &pbuf[wave][l16][0];   // same-wave DS: in-order
        h16v ap = mk16(ld8(prow + halfid * 16), ld8(prow + halfid * 16 + 8));

        // ---- y += P @ V  (V^T layout: K-dim contiguous per column d) ----
#pragma unroll
        for (int dt = 0; dt < 4; ++dt) {
            const _Float16* vp = vt + (bh * HDIM + dt * 16 + l16) * TSEQ + kt + halfid * 16;
            h16v bv = mk16(ld8(vp), ld8(vp + 8));
            vy[dt] = wmma16(ap, bv, vy[dt]);
        }
    }

    // ---- normalize and emit y as f16 [B,T,C] ----
#pragma unroll
    for (int dt = 0; dt < 4; ++dt) {
        int col = h * HDIM + dt * 16 + l16;
#pragma unroll
        for (int i = 0; i < 8; ++i) {
            int rowm = row0 + i + 8 * halfid;
            yout[((size_t)b * TSEQ + rowm) * CDIM + col] = (_Float16)(vy[dt][i] / l_i[i]);
        }
    }
}

// ---------------------------------------------------------------------------
// Host side: stage f16 weights + run the block pipeline
// ---------------------------------------------------------------------------
extern "C" void kernel_launch(void* const* d_in, const int* in_sizes, int n_in,
                              void* d_out, int out_size, void* d_ws, size_t ws_size,
                              hipStream_t stream) {
    const float* x      = (const float*)d_in[0];
    const float* ln1_g  = (const float*)d_in[1];
    const float* ln1_b  = (const float*)d_in[2];
    const float* ln2_g  = (const float*)d_in[3];
    const float* ln2_b  = (const float*)d_in[4];
    const float* qkv_w  = (const float*)d_in[5];
    const float* qkv_b  = (const float*)d_in[6];
    const float* proj_w = (const float*)d_in[7];
    const float* proj_b = (const float*)d_in[8];
    const float* fc1_w  = (const float*)d_in[9];
    const float* fc1_b  = (const float*)d_in[10];
    const float* fc2_w  = (const float*)d_in[11];
    const float* fc2_b  = (const float*)d_in[12];
    float* out = (float*)d_out;

    // workspace carve-up (all chunks 16B aligned)
    _Float16* h1  = (_Float16*)d_ws;                    // LN1(x) f16     [TOK,C]
    _Float16* qf  = h1  + (size_t)TOK * CDIM;           // q,k,v^T f16    3x[TOK,C]
    _Float16* yf  = qf  + (size_t)3 * TOK * CDIM;       // attn out f16   [TOK,C]
    float*    x1  = (float*)(yf + (size_t)TOK * CDIM);  // resid1 f32     [TOK,C]
    _Float16* h2  = (_Float16*)(x1 + (size_t)TOK * CDIM);
    _Float16* gb  = h2  + (size_t)TOK * CDIM;           // gelu(fc1) f16  [TOK,4C]
    _Float16* wq  = gb  + (size_t)TOK * 4 * CDIM;       // f16 weights
    _Float16* wp  = wq  + (size_t)3 * CDIM * CDIM;
    _Float16* wf1 = wp  + (size_t)CDIM * CDIM;
    _Float16* wf2 = wf1 + (size_t)4 * CDIM * CDIM;

    dim3 blk(256);
    f32_to_f16_kernel<<<(3 * CDIM * CDIM + 255) / 256, blk, 0, stream>>>(qkv_w, wq, 3 * CDIM * CDIM);
    f32_to_f16_kernel<<<(CDIM * CDIM + 255) / 256, blk, 0, stream>>>(proj_w, wp, CDIM * CDIM);
    f32_to_f16_kernel<<<(4 * CDIM * CDIM + 255) / 256, blk, 0, stream>>>(fc1_w, wf1, 4 * CDIM * CDIM);
    f32_to_f16_kernel<<<(4 * CDIM * CDIM + 255) / 256, blk, 0, stream>>>(fc2_w, wf2, 4 * CDIM * CDIM);

    // h1 = LN1(x)
    ln_f16_kernel<<<TOK, blk, 0, stream>>>(x, ln1_g, ln1_b, h1);

    // qkv = h1 @ qkv_w^T + b  -> scattered q (scaled), k, v^T in f16
    gemm_wmma_kernel<<<dim3(3 * CDIM / 64, TOK / 256), blk, 0, stream>>>(
        h1, wq, qkv_b, TOK, 3 * CDIM, CDIM, 1, nullptr, nullptr, qf);

    // flash attention
    attn_wmma_kernel<<<NBATCH * NHEAD * (TSEQ / 128), blk, 0, stream>>>(
        qf, qf + (size_t)TOK * CDIM, qf + (size_t)2 * TOK * CDIM, yf);

    // x1 = x + y @ proj_w^T + b
    gemm_wmma_kernel<<<dim3(CDIM / 64, TOK / 256), blk, 0, stream>>>(
        yf, wp, proj_b, TOK, CDIM, CDIM, 0, x, x1, nullptr);

    // h2 = LN2(x1)
    ln_f16_kernel<<<TOK, blk, 0, stream>>>(x1, ln2_g, ln2_b, h2);

    // gb = gelu(h2 @ fc1_w^T + b)
    gemm_wmma_kernel<<<dim3(4 * CDIM / 64, TOK / 256), blk, 0, stream>>>(
        h2, wf1, fc1_b, TOK, 4 * CDIM, CDIM, 2, nullptr, nullptr, gb);

    // out = x1 + gb @ fc2_w^T + b
    gemm_wmma_kernel<<<dim3(CDIM / 64, TOK / 256), blk, 0, stream>>>(
        gb, wf2, fc2_b, TOK, CDIM, 4 * CDIM, 0, x1, out, nullptr);
}
